// EdgeFeatureNet_41489384079743
// MI455X (gfx1250) — matrix operand order repaired and checked by reading, not compiled, not run on hardware
//
#include <hip/hip_runtime.h>
#include <hip/hip_bf16.h>

typedef __attribute__((ext_vector_type(16))) _Float16 v16h;
typedef __attribute__((ext_vector_type(8)))  float    v8f;
typedef __attribute__((ext_vector_type(4)))  float    v4f;

#define NRES   512
#define HID    128
#define NBINS  22
#define PI_F   3.14159265358979323846f

// ---------------------------------------------------------------------------
// Setup kernel 1: n2e[i][o] = node_features[i] @ W_n2e + b_n2e   (512x256x128)
// ---------------------------------------------------------------------------
__global__ void k_n2e(const float* __restrict__ nf, const float* __restrict__ W,
                      const float* __restrict__ b, float* __restrict__ n2e) {
    int i = blockIdx.x, o = threadIdx.x;
    float acc = b[o];
    const float* nfi = nf + i * 256;
    for (int k = 0; k < 256; ++k) acc += nfi[k] * W[k * HID + o];
    n2e[i * HID + o] = acc;
}

// ---------------------------------------------------------------------------
// Setup kernel 2: Arow[i] = n2e[i]@W1[0:128]   + b1 + flow[i]*W1[428]
//                 Acol[i] = n2e[i]@W1[128:256] +      flow[i]*W1[429]
// ---------------------------------------------------------------------------
__global__ void k_rowcol(const float* __restrict__ n2e, const float* __restrict__ W1,
                         const float* __restrict__ b1, const float* __restrict__ flow,
                         float* __restrict__ Arow, float* __restrict__ Acol) {
    int i = blockIdx.x, o = threadIdx.x;
    float ar = b1[o], ac = 0.0f;
    const float* v = n2e + i * HID;
    for (int k = 0; k < HID; ++k) {
        float x = v[k];
        ar += x * W1[k * HID + o];
        ac += x * W1[(HID + k) * HID + o];
    }
    float f = flow[i];
    ar += f * W1[428 * HID + o];
    ac += f * W1[429 * HID + o];
    Arow[i * HID + o] = ar;
    Acol[i * HID + o] = ac;
}

// ---------------------------------------------------------------------------
// Setup kernel 3: rp[d][o] = sinemb(d-511) @ W_rp + b_rp   (1023 x 128)
// ---------------------------------------------------------------------------
__global__ void k_rp(const float* __restrict__ W_rp, const float* __restrict__ b_rp,
                     float* __restrict__ rp) {
    int di = blockIdx.x, o = threadIdx.x;
    float d = (float)(di - 511);
    float acc = b_rp[o];
    const float c0 = (-2.0f / 128.0f) * log2f(2056.0f);
    for (int k = 0; k < 64; ++k) {
        float ang = d * PI_F * exp2f(c0 * (float)k);
        acc += sinf(ang) * W_rp[k * HID + o] + cosf(ang) * W_rp[(64 + k) * HID + o];
    }
    rp[di * HID + o] = acc;
}

// ---------------------------------------------------------------------------
// Setup kernel 4: Rtab[d] = rp[d] @ W1[256:384]
// ---------------------------------------------------------------------------
__global__ void k_rtab(const float* __restrict__ rp, const float* __restrict__ W1,
                       float* __restrict__ Rtab) {
    int di = blockIdx.x, o = threadIdx.x;
    float acc = 0.0f;
    const float* r = rp + di * HID;
    for (int k = 0; k < HID; ++k) acc += r[k] * W1[(256 + k) * HID + o];
    Rtab[di * HID + o] = acc;
}

// ---------------------------------------------------------------------------
// Setup kernel 5: shuffle W2ᵀ / W3ᵀ into WMMA A-fragment order (f16).
// A-frag k order used here (contiguous, same as B): lane L row m=L&15; K = c*32 + (L>>4)*16 + h
// frag element index = ((t*4+c)*32 + L)*16 + h
// ---------------------------------------------------------------------------
__global__ void k_wfrag(const float* __restrict__ W2, const float* __restrict__ W3,
                        _Float16* __restrict__ W2a, _Float16* __restrict__ W3a) {
    int idx = blockIdx.x * 256 + threadIdx.x;   // 2 * 16384
    int which = idx >> 14;
    int rem = idx & 16383;
    int h = rem & 15, L = (rem >> 4) & 31, c = (rem >> 9) & 3, t = (rem >> 11) & 7;
    int k = c * 32 + (L >> 4) * 16 + h;
    int o = t * 16 + (L & 15);
    const float* W = which ? W3 : W2;
    _Float16* dst = which ? W3a : W2a;
    dst[rem] = (_Float16)W[k * HID + o];
}

// ---------------------------------------------------------------------------
// Main kernel: one wave = 16 edges (row i, 16 consecutive j). 64 WMMAs / wave.
// ---------------------------------------------------------------------------
__device__ __forceinline__ void calc_bin(float d, int& bin, float& sc) {
    const float step = (20.0f - 0.001f) / 21.0f;
    int k = (int)floorf((d - 0.001f) / step);
    if (k > 21) k = 21;
    sc = (k >= 0 && d > 0.001f + (float)k * step) ? 1.0f : 0.0f;
    bin = k < 0 ? 0 : k;
}

__global__ __launch_bounds__(256) void k_edge(
    const float* __restrict__ trans, const float* __restrict__ sct,
    const float* __restrict__ emask, const float* __restrict__ W1,
    const float* __restrict__ b2, const float* __restrict__ b3,
    const float* __restrict__ lng, const float* __restrict__ lnb,
    const float* __restrict__ Arow, const float* __restrict__ Acol,
    const float* __restrict__ Rtab,
    const v16h* __restrict__ W2a, const v16h* __restrict__ W3a,
    float* __restrict__ out)
{
    __shared__ __align__(16) float sWd[NBINS * HID];
    __shared__ __align__(16) float sWe[NBINS * HID];
    for (int idx = threadIdx.x; idx < NBINS * HID; idx += 256) {
        sWd[idx] = W1[384 * HID + idx];
        sWe[idx] = W1[406 * HID + idx];
    }
    __syncthreads();

    const int L    = threadIdx.x & 31;
    const int wv   = threadIdx.x >> 5;
    const int g    = blockIdx.x * 8 + wv;       // 0..16383
    const int i    = g >> 5;
    const int jb   = g & 31;
    const int e    = L & 15;
    const int half = L >> 4;
    const int j    = jb * 16 + e;

    // ---- distogram bins for this edge ----
    float dx = trans[i * 3 + 0] - trans[j * 3 + 0];
    float dy = trans[i * 3 + 1] - trans[j * 3 + 1];
    float dz = trans[i * 3 + 2] - trans[j * 3 + 2];
    float dt = sqrtf(dx * dx + dy * dy + dz * dz);
    dx = sct[i * 3 + 0] - sct[j * 3 + 0];
    dy = sct[i * 3 + 1] - sct[j * 3 + 1];
    dz = sct[i * 3 + 2] - sct[j * 3 + 2];
    float ds = sqrtf(dx * dx + dy * dy + dz * dz);
    int bt, bs; float st, ss;
    calc_bin(dt, bt, st);
    calc_bin(ds, bs, ss);

    // ---- layer 1: build h1ᵀ B-fragments directly in registers ----
    // B layout: lane L = edge L&15; K = c*32 + half*16 + h (h = 0..15 contiguous)
    const v4f* ar = (const v4f*)(Arow + i * HID);
    const v4f* ac = (const v4f*)(Acol + j * HID);
    const v4f* rr = (const v4f*)(Rtab + (i - j + 511) * HID);
    const v4f* wd = (const v4f*)(sWd + bt * HID);
    const v4f* we = (const v4f*)(sWe + bs * HID);
    v16h B1[4];
#pragma unroll
    for (int c = 0; c < 4; ++c) {
        int f4 = (c * 32 + half * 16) >> 2;
        v16h bf;
#pragma unroll
        for (int q = 0; q < 4; ++q) {
            v4f v = ar[f4 + q] + ac[f4 + q] + rr[f4 + q] + st * wd[f4 + q] + ss * we[f4 + q];
#pragma unroll
            for (int r = 0; r < 4; ++r) {
                float x = v[r];
                bf[q * 4 + r] = (_Float16)(x > 0.0f ? x : 0.0f);
            }
        }
        B1[c] = bf;
    }

    // ---- layer 2: h2ᵀ = W2ᵀ · h1ᵀ (8 M-tiles × 4 K-chunks) ----
    v8f acc2[8];
#pragma unroll
    for (int t = 0; t < 8; ++t) acc2[t] = (v8f){0.f, 0.f, 0.f, 0.f, 0.f, 0.f, 0.f, 0.f};
#pragma unroll
    for (int t = 0; t < 8; ++t) {
#pragma unroll
        for (int c = 0; c < 4; ++c) {
            v16h A = W2a[(t * 4 + c) * 32 + L];
            acc2[t] = __builtin_amdgcn_wmma_f32_16x16x32_f16(
                false, A, false, B1[c], (short)0, acc2[t], false, false);
        }
    }

    // ---- bias + relu + f16 pack; D-frag -> next-layer B-frag via shfl_xor(16) ----
    unsigned int packed[8][4];
#pragma unroll
    for (int t = 0; t < 8; ++t) {
        int m0 = t * 16 + half * 8;
        v4f ba = *(const v4f*)(b2 + m0);
        v4f bb = *(const v4f*)(b2 + m0 + 4);
        _Float16 h16[8];
#pragma unroll
        for (int r = 0; r < 4; ++r) {
            float x = acc2[t][r] + ba[r];     h16[r]     = (_Float16)(x > 0.f ? x : 0.f);
            float y = acc2[t][4 + r] + bb[r]; h16[4 + r] = (_Float16)(y > 0.f ? y : 0.f);
        }
#pragma unroll
        for (int w = 0; w < 4; ++w) {
            union { _Float16 h[2]; unsigned int u; } cv;
            cv.h[0] = h16[2 * w]; cv.h[1] = h16[2 * w + 1];
            packed[t][w] = cv.u;
        }
    }
    v16h B2[4];
#pragma unroll
    for (int c = 0; c < 4; ++c) {
        union { unsigned int u[8]; v16h v; } cv;
#pragma unroll
        for (int w = 0; w < 4; ++w) {
            unsigned int send = packed[2 * c + 1 - half][w];
            unsigned int recv = (unsigned int)__shfl_xor((int)send, 16, 32);
            unsigned int own  = packed[2 * c + half][w];
            cv.u[w]     = half ? recv : own;
            cv.u[4 + w] = half ? own  : recv;
        }
        B2[c] = cv.v;
    }

    // ---- layer 3: h3ᵀ = W3ᵀ · h2ᵀ ----
    v8f acc3[8];
#pragma unroll
    for (int t = 0; t < 8; ++t) acc3[t] = (v8f){0.f, 0.f, 0.f, 0.f, 0.f, 0.f, 0.f, 0.f};
#pragma unroll
    for (int t = 0; t < 8; ++t) {
#pragma unroll
        for (int c = 0; c < 4; ++c) {
            v16h A = W3a[(t * 4 + c) * 32 + L];
            acc3[t] = __builtin_amdgcn_wmma_f32_16x16x32_f16(
                false, A, false, B2[c], (short)0, acc3[t], false, false);
        }
    }

    // ---- LayerNorm over 128 features per edge + mask + store ----
    float s1 = 0.f, s2 = 0.f;
#pragma unroll
    for (int t = 0; t < 8; ++t) {
        int m0 = t * 16 + half * 8;
        v4f ba = *(const v4f*)(b3 + m0);
        v4f bb = *(const v4f*)(b3 + m0 + 4);
#pragma unroll
        for (int r = 0; r < 4; ++r) {
            float x = acc3[t][r] + ba[r];     s1 += x; s2 += x * x;
            float y = acc3[t][4 + r] + bb[r]; s1 += y; s2 += y * y;
        }
    }
    s1 += __shfl_xor(s1, 16, 32);
    s2 += __shfl_xor(s2, 16, 32);
    float mu  = s1 * (1.0f / 128.0f);
    float var = s2 * (1.0f / 128.0f) - mu * mu;
    float rs  = rsqrtf(var + 1e-5f);
    float mk  = emask[i * NRES + j];
    float* ob = out + ((size_t)(i * NRES + j)) * HID;
#pragma unroll
    for (int t = 0; t < 8; ++t) {
        int m0 = t * 16 + half * 8;
        v4f ba = *(const v4f*)(b3 + m0);
        v4f bb = *(const v4f*)(b3 + m0 + 4);
        v4f ga = *(const v4f*)(lng + m0);
        v4f gb = *(const v4f*)(lng + m0 + 4);
        v4f la = *(const v4f*)(lnb + m0);
        v4f lb = *(const v4f*)(lnb + m0 + 4);
        v4f o1, o2;
#pragma unroll
        for (int r = 0; r < 4; ++r) {
            float x = acc3[t][r] + ba[r];
            float y = acc3[t][4 + r] + bb[r];
            o1[r] = ((x - mu) * rs * ga[r] + la[r]) * mk;
            o2[r] = ((y - mu) * rs * gb[r] + lb[r]) * mk;
        }
        *(v4f*)(ob + m0)     = o1;
        *(v4f*)(ob + m0 + 4) = o2;
    }
}

// ---------------------------------------------------------------------------
extern "C" void kernel_launch(void* const* d_in, const int* in_sizes, int n_in,
                              void* d_out, int out_size, void* d_ws, size_t ws_size,
                              hipStream_t stream) {
    const float* nf    = (const float*)d_in[0];
    const float* trans = (const float*)d_in[1];
    const float* sct   = (const float*)d_in[2];
    const float* emask = (const float*)d_in[3];
    const float* flow  = (const float*)d_in[4];
    const float* W_n2e = (const float*)d_in[5];
    const float* b_n2e = (const float*)d_in[6];
    const float* W_rp  = (const float*)d_in[7];
    const float* b_rp  = (const float*)d_in[8];
    const float* W1    = (const float*)d_in[9];
    const float* b1    = (const float*)d_in[10];
    const float* W2    = (const float*)d_in[11];
    const float* b2    = (const float*)d_in[12];
    const float* W3    = (const float*)d_in[13];
    const float* b3    = (const float*)d_in[14];
    const float* lng   = (const float*)d_in[15];
    const float* lnb   = (const float*)d_in[16];

    float* ws   = (float*)d_ws;
    float* n2e  = ws;               // 512*128            = 65536
    float* Arow = ws + 65536;       // 65536
    float* Acol = ws + 131072;      // 65536
    float* rp   = ws + 196608;      // 1023*128           = 130944
    float* Rtab = ws + 327552;      // 130944
    _Float16* W2a = (_Float16*)(ws + 458496);  // 16384 halves (32B aligned)
    _Float16* W3a = W2a + 16384;               // 16384 halves

    k_n2e   <<<512, 128, 0, stream>>>(nf, W_n2e, b_n2e, n2e);
    k_rowcol<<<512, 128, 0, stream>>>(n2e, W1, b1, flow, Arow, Acol);
    k_rp    <<<1023, 128, 0, stream>>>(W_rp, b_rp, rp);
    k_rtab  <<<1023, 128, 0, stream>>>(rp, W1, Rtab);
    k_wfrag <<<128, 256, 0, stream>>>(W2, W3, W2a, W3a);
    k_edge  <<<2048, 256, 0, stream>>>(trans, sct, emask, W1, b2, b3, lng, lnb,
                                       Arow, Acol, Rtab,
                                       (const v16h*)W2a, (const v16h*)W3a,
                                       (float*)d_out);
}
